// MultiHeadAttention_70016556859683
// MI455X (gfx1250) — matrix operand order repaired and checked by reading, not compiled, not run on hardware
//
#include <hip/hip_runtime.h>
#include <hip/hip_bf16.h>

// MHA block for B=2, T=2048, E=1024, H=16, D=64 on gfx1250 (wave32, WMMA).
// Outputs: [0 .. B*T*E)        : layernorm(query + proj)   (f32)
//          [B*T*E .. +B*H*T*T) : attn_weights              (f32)
//
// v2: all WMMA fragment loads are contiguous 32B/lane (weights + V transposed),
//     each wave computes a 16x64 strip (4 wmma per A-fragment load).

#define BDIM 2
#define TDIM 2048
#define EDIM 1024
#define HDS  16
#define DDIM 64
#define LN_EPS 1e-5f

typedef __attribute__((ext_vector_type(16))) _Float16 v16h;
typedef __attribute__((ext_vector_type(8)))  float    v8f;

// ---------------------------------------------------------------- conversions
__global__ __launch_bounds__(256) void cvt_f32_to_f16(const float* __restrict__ in,
                                                      _Float16* __restrict__ out, int n) {
  int i = blockIdx.x * 256 + threadIdx.x;
  if (i < n) out[i] = (_Float16)in[i];
}

// out[n*K + k] = (f16) in[k*N + n]   (LDS-tiled 32x32 transpose, 32x8 threads)
__global__ __launch_bounds__(256) void transpose_cvt_f32_to_f16(
    const float* __restrict__ in, _Float16* __restrict__ out, int K, int N) {
  __shared__ _Float16 t[32][33];
  const int tx = threadIdx.x & 31;
  const int ty = threadIdx.x >> 5;            // 0..7
  const int n0 = blockIdx.x << 5;
  const int k0 = blockIdx.y << 5;
#pragma unroll
  for (int j = 0; j < 32; j += 8)
    t[ty + j][tx] = (_Float16)in[(size_t)(k0 + ty + j) * N + n0 + tx];
  __syncthreads();
#pragma unroll
  for (int j = 0; j < 32; j += 8)
    out[(size_t)(n0 + ty + j) * K + k0 + tx] = t[tx][ty + j];
}

// ---------------------------------------------------------------- generic GEMM
// C[M,N] = A[M,K] (f16 row-major) * B[K,N] + bias[N], with B supplied
// TRANSPOSED (Bt is N x K row-major) so fragment loads are contiguous.
// One 16(M) x 64(N) strip per wave; K-loop step 32; 4 wmma per iteration.
// OUT_MODE: 0 = f32 row-major, 1 = f16 row-major, 2 = f16 (B,E,T) transposed.
template <int OUT_MODE>
__global__ __launch_bounds__(256) void gemm_wmma(
    const _Float16* __restrict__ A, const _Float16* __restrict__ Bt,
    const float* __restrict__ bias, _Float16* __restrict__ outH,
    float* __restrict__ outF, int Mdim, int Ndim, int Kdim) {
  const int wave = blockIdx.x * (blockDim.x >> 5) + (threadIdx.x >> 5);
  const int lane = threadIdx.x & 31;
  const int nstrips = Ndim >> 6;
  if (wave >= (Mdim >> 4) * nstrips) return;
  const int m0 = (wave / nstrips) << 4;
  const int n0 = (wave % nstrips) << 6;

  // A fragment (16x32 f16): row = m0 + lane%16; lanes 0-15: K 0-15,
  // lanes 16-31: K 16-31.
  const int ar  = m0 + (lane & 15);
  const int akb = (lane >> 4) << 4;
  // B fragment (32x16): col = lane%16; lanes 0-15: K 0-15, lanes 16-31: K 16-31.
  const int bcl = lane & 15;
  const int bkb = (lane >> 4) << 4;

  v8f acc[4] = {};
  for (int k0 = 0; k0 < Kdim; k0 += 32) {
    v16h a;
    const _Float16* ap = A + (size_t)ar * Kdim + (k0 + akb);
#pragma unroll
    for (int j = 0; j < 8; ++j) { a[j] = ap[j]; a[8 + j] = ap[8 + j]; }
#pragma unroll
    for (int t = 0; t < 4; ++t) {
      v16h bf;
      const _Float16* bp = Bt + (size_t)(n0 + 16 * t + bcl) * Kdim + (k0 + bkb);
#pragma unroll
      for (int e = 0; e < 16; ++e) bf[e] = bp[e];
      acc[t] = __builtin_amdgcn_wmma_f32_16x16x32_f16(false, a, false, bf,
                                                      (short)0, acc[t], false, false);
    }
  }
  const int rbase = m0 + ((lane >> 4) << 3);   // C/D: VGPR r -> row r / r+8
#pragma unroll
  for (int t = 0; t < 4; ++t) {
    const int col = n0 + 16 * t + bcl;
    const float bv = bias[col];
#pragma unroll
    for (int r = 0; r < 8; ++r) {
      float v = acc[t][r] + bv;
      if (OUT_MODE == 0) {
        outF[(size_t)(rbase + r) * Ndim + col] = v;
      } else if (OUT_MODE == 1) {
        outH[(size_t)(rbase + r) * Ndim + col] = (_Float16)v;
      } else {  // (B, E, T) transposed f16 (for V)
        const int row = rbase + r;
        const int b_  = row / TDIM;
        const int tt  = row - b_ * TDIM;
        outH[((size_t)b_ * Ndim + col) * TDIM + tt] = (_Float16)v;
      }
    }
  }
}

// ---------------------------------------------------------------- scores QK^T
// One wave per 16(q) x 64(k) strip; 2 d-steps x 4 wmma.
__global__ __launch_bounds__(256) void scores_wmma_kernel(
    const _Float16* __restrict__ Q16, const _Float16* __restrict__ K16,
    const int* __restrict__ mask, float* __restrict__ attn) {
  const int wave = blockIdx.x * (blockDim.x >> 5) + (threadIdx.x >> 5);
  const int lane = threadIdx.x & 31;
  const int tpq = TDIM >> 4;                  // 128 q-tiles
  const int tpk = TDIM >> 6;                  // 32 k-strips
  const int tph = tpq * tpk;
  const int bh  = wave / tph;
  const int rem = wave % tph;
  const int q0  = (rem / tpk) << 4;
  const int kt0 = (rem % tpk) << 6;
  const int b = bh / HDS, h = bh % HDS;

  const _Float16* Qb = Q16 + (size_t)b * TDIM * EDIM + h * DDIM;
  const _Float16* Kb = K16 + (size_t)b * TDIM * EDIM + h * DDIM;

  const int aq  = q0 + (lane & 15);
  const int akb = (lane >> 4) << 4;
  const int bcl = lane & 15;
  const int bkb = (lane >> 4) << 4;

  v8f acc[4] = {};
#pragma unroll
  for (int d0 = 0; d0 < DDIM; d0 += 32) {
    v16h a;
    const _Float16* ap = Qb + (size_t)aq * EDIM + d0 + akb;
#pragma unroll
    for (int j = 0; j < 8; ++j) { a[j] = ap[j]; a[8 + j] = ap[8 + j]; }
#pragma unroll
    for (int t = 0; t < 4; ++t) {
      v16h bf;   // B[d,k] = K[k,d] -> contiguous read of K row
      const _Float16* bp = Kb + (size_t)(kt0 + 16 * t + bcl) * EDIM + d0 + bkb;
#pragma unroll
      for (int e = 0; e < 16; ++e) bf[e] = bp[e];
      acc[t] = __builtin_amdgcn_wmma_f32_16x16x32_f16(false, a, false, bf,
                                                      (short)0, acc[t], false, false);
    }
  }
  const float scale = 0.125f;                 // 1/sqrt(64)
  const int rbase = q0 + ((lane >> 4) << 3);
  float* Sout = attn + (size_t)bh * TDIM * TDIM;
  const int* Mb = mask + (size_t)b * TDIM * TDIM;   // (B,1,T,T)
#pragma unroll
  for (int t = 0; t < 4; ++t) {
    const int col = kt0 + 16 * t + bcl;
#pragma unroll
    for (int r = 0; r < 8; ++r) {
      const int q = rbase + r;
      float v = acc[t][r] * scale;
      if (Mb[(size_t)q * TDIM + col] == 0) v = -__builtin_inff();
      Sout[(size_t)q * TDIM + col] = v;
    }
  }
}

// ---------------------------------------------------------------- row softmax
__global__ __launch_bounds__(256) void softmax_rows(float* __restrict__ attn) {
  __shared__ float red[256];
  float* p = attn + (size_t)blockIdx.x * TDIM;
  const int tid = threadIdx.x;
  float mx = -__builtin_inff();
  for (int i = tid; i < TDIM; i += 256) mx = fmaxf(mx, p[i]);
  red[tid] = mx; __syncthreads();
  for (int s = 128; s > 0; s >>= 1) { if (tid < s) red[tid] = fmaxf(red[tid], red[tid + s]); __syncthreads(); }
  mx = red[0]; __syncthreads();
  float sum = 0.f;
  for (int i = tid; i < TDIM; i += 256) { float e = __expf(p[i] - mx); p[i] = e; sum += e; }
  red[tid] = sum; __syncthreads();
  for (int s = 128; s > 0; s >>= 1) { if (tid < s) red[tid] += red[tid + s]; __syncthreads(); }
  const float inv = 1.0f / red[0];
  for (int i = tid; i < TDIM; i += 256) p[i] *= inv;
}

// ---------------------------------------------------------------- attn @ V
// Vt layout (B, E, T) so B-fragments are contiguous. One wave computes a
// 16(t) x 64(d) strip = the full head dim; 4 wmma per A-fragment conversion.
__global__ __launch_bounds__(256) void attnout_wmma_kernel(
    const float* __restrict__ attn, const _Float16* __restrict__ Vt,
    _Float16* __restrict__ C16) {
  const int wave = blockIdx.x * (blockDim.x >> 5) + (threadIdx.x >> 5);
  const int lane = threadIdx.x & 31;
  const int tph = TDIM >> 4;                  // 128 t-tiles per (b,h)
  const int bh = wave / tph;
  const int t0 = (wave % tph) << 4;
  const int b = bh / HDS, h = bh % HDS;

  const float*    Pb = attn + (size_t)bh * TDIM * TDIM;
  const _Float16* Vb = Vt + ((size_t)b * EDIM + h * DDIM) * TDIM;

  const int at  = t0 + (lane & 15);
  const int akb = (lane >> 4) << 4;
  const int bcl = lane & 15;
  const int bkb = (lane >> 4) << 4;

  v8f acc[4] = {};
  for (int k0 = 0; k0 < TDIM; k0 += 32) {
    v16h a;
    const float* ap = Pb + (size_t)at * TDIM + k0 + akb;
#pragma unroll
    for (int j = 0; j < 8; ++j) { a[j] = (_Float16)ap[j]; a[8 + j] = (_Float16)ap[8 + j]; }
#pragma unroll
    for (int t = 0; t < 4; ++t) {
      v16h bf;   // B[k,d] = Vt[(d row) * T + k] -> contiguous in k
      const _Float16* bp = Vb + (size_t)(16 * t + bcl) * TDIM + k0 + bkb;
#pragma unroll
      for (int e = 0; e < 16; ++e) bf[e] = bp[e];
      acc[t] = __builtin_amdgcn_wmma_f32_16x16x32_f16(false, a, false, bf,
                                                      (short)0, acc[t], false, false);
    }
  }
  _Float16* Cb = C16 + (size_t)b * TDIM * EDIM + h * DDIM;
  const int rbase = t0 + ((lane >> 4) << 3);
#pragma unroll
  for (int t = 0; t < 4; ++t) {
    const int col = 16 * t + bcl;
#pragma unroll
    for (int r = 0; r < 8; ++r)
      Cb[(size_t)(rbase + r) * EDIM + col] = (_Float16)acc[t][r];
  }
}

// ---------------------------------------------------------------- residual + LN
__global__ __launch_bounds__(256) void add_layernorm(
    const float* __restrict__ x, const float* __restrict__ proj,
    const float* __restrict__ gamma, const float* __restrict__ beta,
    float* __restrict__ out) {
  __shared__ float red[256];
  const int tid = threadIdx.x;
  const float* xp = x + (size_t)blockIdx.x * EDIM;
  const float* pp = proj + (size_t)blockIdx.x * EDIM;
  float vals[4];
  float s = 0.f;
#pragma unroll
  for (int j = 0; j < 4; ++j) { vals[j] = xp[tid + 256 * j] + pp[tid + 256 * j]; s += vals[j]; }
  red[tid] = s; __syncthreads();
  for (int st = 128; st > 0; st >>= 1) { if (tid < st) red[tid] += red[tid + st]; __syncthreads(); }
  const float mean = red[0] * (1.0f / EDIM);
  __syncthreads();
  float sq = 0.f;
#pragma unroll
  for (int j = 0; j < 4; ++j) { float d = vals[j] - mean; sq += d * d; }
  red[tid] = sq; __syncthreads();
  for (int st = 128; st > 0; st >>= 1) { if (tid < st) red[tid] += red[tid + st]; __syncthreads(); }
  const float inv = rsqrtf(red[0] * (1.0f / EDIM) + LN_EPS);
  float* op = out + (size_t)blockIdx.x * EDIM;
#pragma unroll
  for (int j = 0; j < 4; ++j) {
    int i = tid + 256 * j;
    op[i] = (vals[j] - mean) * inv * gamma[i] + beta[i];
  }
}

// ---------------------------------------------------------------- launch
extern "C" void kernel_launch(void* const* d_in, const int* in_sizes, int n_in,
                              void* d_out, int out_size, void* d_ws, size_t ws_size,
                              hipStream_t stream) {
  (void)in_sizes; (void)n_in; (void)out_size; (void)ws_size;
  const float* query = (const float*)d_in[0];
  const int*   mask  = (const int*)d_in[1];
  const float* wq = (const float*)d_in[2];
  const float* bq = (const float*)d_in[3];
  const float* wk = (const float*)d_in[4];
  const float* bk = (const float*)d_in[5];
  const float* wv = (const float*)d_in[6];
  const float* bv = (const float*)d_in[7];
  const float* wo = (const float*)d_in[8];
  const float* bo = (const float*)d_in[9];
  const float* gamma = (const float*)d_in[10];
  const float* beta  = (const float*)d_in[11];

  float* out  = (float*)d_out;
  float* attn = out + (size_t)BDIM * TDIM * EDIM;   // attn_weights region

  // workspace carve-up (~64 MB)
  char* ws = (char*)d_ws;
  size_t off = 0;
  auto carve = [&](size_t bytes) { char* p = ws + off; off += (bytes + 255) & ~(size_t)255; return p; };
  const size_t nX = (size_t)BDIM * TDIM * EDIM;     // 4M elems
  const size_t nW = (size_t)EDIM * EDIM;            // 1M elems
  _Float16* X16   = (_Float16*)carve(nX * 2);
  _Float16* WqT16 = (_Float16*)carve(nW * 2);       // transposed (out,in)
  _Float16* WkT16 = (_Float16*)carve(nW * 2);
  _Float16* WvT16 = (_Float16*)carve(nW * 2);
  _Float16* WoT16 = (_Float16*)carve(nW * 2);
  _Float16* Q16   = (_Float16*)carve(nX * 2);       // (B,T,E)
  _Float16* K16   = (_Float16*)carve(nX * 2);       // (B,T,E)
  _Float16* Vt16  = (_Float16*)carve(nX * 2);       // (B,E,T) transposed
  _Float16* C16   = (_Float16*)carve(nX * 2);       // concat (B,T,E)
  float*    Pf    = (float*)carve(nX * 4);          // proj (f32)

  // staging: query f16, weights transposed f16
  cvt_f32_to_f16<<<(int)(nX / 256), 256, 0, stream>>>(query, X16, (int)nX);
  dim3 tgrid(EDIM / 32, EDIM / 32);
  transpose_cvt_f32_to_f16<<<tgrid, 256, 0, stream>>>(wq, WqT16, EDIM, EDIM);
  transpose_cvt_f32_to_f16<<<tgrid, 256, 0, stream>>>(wk, WkT16, EDIM, EDIM);
  transpose_cvt_f32_to_f16<<<tgrid, 256, 0, stream>>>(wv, WvT16, EDIM, EDIM);
  transpose_cvt_f32_to_f16<<<tgrid, 256, 0, stream>>>(wo, WoT16, EDIM, EDIM);

  // QKV projections: M=4096, N=K=1024 -> (M/16)*(N/64)=4096 waves -> 512 blocks
  const int M = BDIM * TDIM;
  const int projBlocks = (M >> 4) * (EDIM >> 6) / 8;
  gemm_wmma<1><<<projBlocks, 256, 0, stream>>>(X16, WqT16, bq, Q16, nullptr, M, EDIM, EDIM);
  gemm_wmma<1><<<projBlocks, 256, 0, stream>>>(X16, WkT16, bk, K16, nullptr, M, EDIM, EDIM);
  gemm_wmma<2><<<projBlocks, 256, 0, stream>>>(X16, WvT16, bv, Vt16, nullptr, M, EDIM, EDIM);

  // scores: B*H * 128 q-tiles * 32 k-strips = 131072 waves -> 16384 blocks
  const int scoreWaves = BDIM * HDS * (TDIM >> 4) * (TDIM >> 6);
  scores_wmma_kernel<<<scoreWaves / 8, 256, 0, stream>>>(Q16, K16, mask, attn);

  // softmax in place (this region IS the attn_weights output)
  softmax_rows<<<BDIM * HDS * TDIM, 256, 0, stream>>>(attn);

  // attn @ V: B*H * 128 t-tiles = 4096 waves -> 512 blocks
  const int aoWaves = BDIM * HDS * (TDIM >> 4);
  attnout_wmma_kernel<<<aoWaves / 8, 256, 0, stream>>>(attn, Vt16, C16);

  // output projection (f32 out)
  gemm_wmma<0><<<projBlocks, 256, 0, stream>>>(C16, WoT16, bo, nullptr, Pf, M, EDIM, EDIM);

  // residual + layernorm
  add_layernorm<<<BDIM * TDIM, 256, 0, stream>>>(query, Pf, gamma, beta, out);
}
